// DiscreteDeltaThetaGammaLayer_16286515986766
// MI455X (gfx1250) — matrix operand order repaired and checked by reading, not compiled, not run on hardware
//
#include <hip/hip_runtime.h>
#include <hip/hip_bf16.h>
#include <math.h>

#define N_DELTA 32
#define N_THETA 64
#define N_GAMMA 256
#define N_TOTAL 352
#define N_DIMS  1024
#define BATCH   4096
#define N_STEPS 32
#define DT      0.01f
#define COUPLING 2.0f
#define PAC     0.3f
#define EPS     1e-6f
#define TWO_PI      6.2831853071795864769f
#define INV_TWO_PI  0.15915494309189533577f

typedef __attribute__((ext_vector_type(16))) _Float16 v16h;
typedef __attribute__((ext_vector_type(8)))  float    v8f;

__device__ __forceinline__ float wrap2pi(float v) {
    // v mod 2pi, result in [0, 2pi) -- matches Python %
    float r = v * INV_TWO_PI;
    r = r - floorf(r);
    return r * TWO_PI;
}

__device__ __forceinline__ float wredsum(float v) {
    #pragma unroll
    for (int off = 16; off > 0; off >>= 1)
        v += __shfl_xor(v, off, 32);
    return v;
}

// ---------------------------------------------------------------------------
// Kernel 1: P0 = wrap2pi(x @ W_phase^T), A0 = max(|x @ W_amp^T|, EPS)
// One wave = one 16x16 output tile. which=0 -> phase, which=1 -> amp.
// ---------------------------------------------------------------------------
__global__ void __launch_bounds__(256)
init_gemm_kernel(const float* __restrict__ x,
                 const float* __restrict__ Wp,
                 const float* __restrict__ Wa,
                 float* __restrict__ P0,
                 float* __restrict__ A0)
{
    const int lane = threadIdx.x & 31;
    const int wid  = (blockIdx.x * blockDim.x + threadIdx.x) >> 5;

    const int NT = N_TOTAL / 16;          // 22
    const int MT = BATCH / 16;            // 256
    const int tiles_per = MT * NT;        // 5632
    const int which = wid / tiles_per;    // 0: phase, 1: amp
    int rem = wid - which * tiles_per;
    const int mtile = rem / NT;
    const int ntile = rem - mtile * NT;

    const float* __restrict__ W = which ? Wa : Wp;
    const int lm = lane & 15;             // M (for A) / N (for B) within tile
    const int g  = lane >> 4;             // lane group selects K-half

    const float* __restrict__ xrow = x + (size_t)(mtile * 16 + lm) * N_DIMS;
    const float* __restrict__ wrow = W + (size_t)(ntile * 16 + lm) * N_DIMS;

    v8f acc = {};
    #pragma unroll 4
    for (int kb = 0; kb < N_DIMS; kb += 32) {
        // A fragment (16-bit A 16x32 layout): group g holds K in
        // [8g,8g+8) and [16+8g,16+8g+8) -> two contiguous runs of 8 floats
        const float4 a0 = *(const float4*)(xrow + kb + 8 * g);
        const float4 a1 = *(const float4*)(xrow + kb + 8 * g + 4);
        const float4 a2 = *(const float4*)(xrow + kb + 16 + 8 * g);
        const float4 a3 = *(const float4*)(xrow + kb + 16 + 8 * g + 4);
        v16h A;
        A[0]=(_Float16)a0.x; A[1]=(_Float16)a0.y; A[2]=(_Float16)a0.z; A[3]=(_Float16)a0.w;
        A[4]=(_Float16)a1.x; A[5]=(_Float16)a1.y; A[6]=(_Float16)a1.z; A[7]=(_Float16)a1.w;
        A[8]=(_Float16)a2.x; A[9]=(_Float16)a2.y; A[10]=(_Float16)a2.z; A[11]=(_Float16)a2.w;
        A[12]=(_Float16)a3.x; A[13]=(_Float16)a3.y; A[14]=(_Float16)a3.z; A[15]=(_Float16)a3.w;

        // B fragment (32x16, B[k][n] = W[n][k]): group g holds K in
        // [8g,8g+8) and [16+8g,16+8g+8) of row n of W (same order as A)
        const float4 b0 = *(const float4*)(wrow + kb + 8 * g);
        const float4 b1 = *(const float4*)(wrow + kb + 8 * g + 4);
        const float4 b2 = *(const float4*)(wrow + kb + 16 + 8 * g);
        const float4 b3 = *(const float4*)(wrow + kb + 16 + 8 * g + 4);
        v16h B;
        B[0]=(_Float16)b0.x; B[1]=(_Float16)b0.y; B[2]=(_Float16)b0.z; B[3]=(_Float16)b0.w;
        B[4]=(_Float16)b1.x; B[5]=(_Float16)b1.y; B[6]=(_Float16)b1.z; B[7]=(_Float16)b1.w;
        B[8]=(_Float16)b2.x; B[9]=(_Float16)b2.y; B[10]=(_Float16)b2.z; B[11]=(_Float16)b2.w;
        B[12]=(_Float16)b3.x; B[13]=(_Float16)b3.y; B[14]=(_Float16)b3.z; B[15]=(_Float16)b3.w;

        acc = __builtin_amdgcn_wmma_f32_16x16x32_f16(
            /*neg_a=*/false, A, /*neg_b=*/false, B,
            /*c_mod=*/(short)0, acc, /*reuse_a=*/false, /*reuse_b=*/false);
    }

    // C/D layout: VGPR j, lanes 0-15 -> M=j, lanes 16-31 -> M=8+j; N=lane&15
    float* __restrict__ out = which ? A0 : P0;
    const int row0 = mtile * 16 + 8 * g;
    const int n    = ntile * 16 + lm;
    #pragma unroll
    for (int j = 0; j < 8; ++j) {
        float v = acc[j];
        v = which ? fmaxf(fabsf(v), EPS) : wrap2pi(v);
        out[(size_t)(row0 + j) * N_TOTAL + n] = v;
    }
}

// ---------------------------------------------------------------------------
// Kernel 2: 32 Kuramoto steps. One wave per batch row; lane l, slot j holds
// oscillator 32*j + l, so blocks (32/64/256) align to slot boundaries:
//   slot 0 = delta, slots 1-2 = theta, slots 3-10 = gamma.
// K is block-diagonal-constant -> coupling = per-block mean-field sums.
// ---------------------------------------------------------------------------
__global__ void __launch_bounds__(256)
dynamics_kernel(const float* __restrict__ P0,
                const float* __restrict__ A0,
                const float* __restrict__ omega,
                float* __restrict__ out)
{
    const int lane = threadIdx.x & 31;
    const int row  = (blockIdx.x * blockDim.x + threadIdx.x) >> 5;

    const size_t base = (size_t)row * N_TOTAL + lane;

    float phase[11], amp[11], om[11];
    #pragma unroll
    for (int j = 0; j < 11; ++j) {
        phase[j] = P0[base + 32 * j];
        amp[j]   = A0[base + 32 * j];
        om[j]    = omega[32 * j + lane];
    }

    const float kd = COUPLING / (float)N_DELTA;
    const float kt = COUPLING / (float)N_THETA;
    const float kg = COUPLING / (float)N_GAMMA;

    for (int step = 0; step < N_STEPS; ++step) {
        float s[11], c[11];
        #pragma unroll
        for (int j = 0; j < 11; ++j) __sincosf(phase[j], &s[j], &c[j]);

        float gs = 0.f, gc = 0.f;
        #pragma unroll
        for (int j = 3; j < 11; ++j) { gs += s[j]; gc += c[j]; }
        const float Sd_s = wredsum(s[0]);
        const float Sd_c = wredsum(c[0]);
        const float St_s = wredsum(s[1] + s[2]);
        const float St_c = wredsum(c[1] + c[2]);
        const float Sg_s = wredsum(gs);
        const float Sg_c = wredsum(gc);

        float pn[11];
        #pragma unroll
        for (int j = 0; j < 11; ++j) {
            float Ss, Sc, kb;
            if (j == 0)     { Ss = Sd_s; Sc = Sd_c; kb = kd; }
            else if (j < 3) { Ss = St_s; Sc = St_c; kb = kt; }
            else            { Ss = Sg_s; Sc = Sg_c; kb = kg; }
            const float coup = c[j] * (kb * Ss) - s[j] * (kb * Sc);
            pn[j] = wrap2pi(phase[j] + DT * (om[j] + coup));
        }

        // circular means of the NEW delta / theta phases
        float snd, cnd; __sincosf(pn[0], &snd, &cnd);
        const float Nd_s = wredsum(snd);
        const float Nd_c = wredsum(cnd);
        float sn1, cn1, sn2, cn2;
        __sincosf(pn[1], &sn1, &cn1);
        __sincosf(pn[2], &sn2, &cn2);
        const float Nt_s = wredsum(sn1 + sn2);
        const float Nt_c = wredsum(cn1 + cn2);

        const float mean_d = atan2f(Nd_s, Nd_c);
        const float mean_t = atan2f(Nt_s, Nt_c);
        const float mod_theta = PAC * __cosf(mean_d);   // theta modulated by delta mean
        const float mod_gamma = PAC * __cosf(mean_t);   // gamma modulated by theta mean

        #pragma unroll
        for (int j = 0; j < 11; ++j) {
            const float mod = (j == 0) ? 0.f : (j < 3 ? mod_theta : mod_gamma);
            amp[j] = fmaxf(amp[j] * (1.f + DT * mod), EPS);
            phase[j] = pn[j];
        }
    }

    #pragma unroll
    for (int j = 0; j < 11; ++j)
        out[base + 32 * j] = amp[j];
}

// ---------------------------------------------------------------------------

extern "C" void kernel_launch(void* const* d_in, const int* in_sizes, int n_in,
                              void* d_out, int out_size, void* d_ws, size_t ws_size,
                              hipStream_t stream) {
    const float* x     = (const float*)d_in[0];   // (4096, 1024)
    const float* Wp    = (const float*)d_in[1];   // (352, 1024)
    const float* Wa    = (const float*)d_in[2];   // (352, 1024)
    const float* omega = (const float*)d_in[3];   // (352,)
    // d_in[4] = K: block-diagonal constant; structure folded into kernel 2.

    float* P0 = (float*)d_ws;                          // (4096, 352) wrapped phases
    float* A0 = P0 + (size_t)BATCH * N_TOTAL;          // (4096, 352) initial amps
    float* out = (float*)d_out;                        // (4096, 352) final amps

    // GEMM: 2 outputs * 256 M-tiles * 22 N-tiles = 11264 waves, 8 waves/block
    const int tiles = 2 * (BATCH / 16) * (N_TOTAL / 16);
    init_gemm_kernel<<<dim3(tiles / 8), dim3(256), 0, stream>>>(x, Wp, Wa, P0, A0);

    // Dynamics: one wave per row, 8 rows per 256-thread block
    dynamics_kernel<<<dim3(BATCH / 8), dim3(256), 0, stream>>>(P0, A0, omega, out);
}
